// RackioEnsembleLSTMCell_48352741819078
// MI455X (gfx1250) — matrix operand order repaired and checked by reading, not compiled, not run on hardware
//
#include <hip/hip_runtime.h>

// ---------------------------------------------------------------------------
// LSTM (Keras gate order i,f,g,o; linear candidate; h = o*c; out = tanh(h_T))
// B=128, T=1024, F=128, U=256, 4U=1024.
// 8 workgroups (one per 16-batch-row group, independent recurrences), 16 waves
// each; wave w owns unit block [16w,16w+16) and computes its 4 gate tiles with
// v_wmma_f32_16x16x32_bf16. c stays in registers, h lives in LDS (bf16,
// A-fragment friendly). W/R pre-packed once to bf16 WMMA B-fragment order in
// workspace (768 KB, L2-resident) and streamed from L2 every step.
// k-chunk loops are deliberately NOT unrolled so the compiler cannot hoist the
// (t-invariant) B-fragment loads and spill them to scratch.
// ---------------------------------------------------------------------------

typedef __bf16 bf16_t;
typedef __attribute__((ext_vector_type(16))) __bf16 v16bf;
typedef __attribute__((ext_vector_type(8)))  __bf16 v8bf;
typedef __attribute__((ext_vector_type(8)))  float  v8f;
typedef __attribute__((ext_vector_type(4)))  float  f32x4;

#define B_TOT 128
#define T_LEN 1024
#define F_IN  128
#define U_DIM 256
#define G4    1024

#define HP 264   // LDS pitch (bf16) for h rows: 256 + 8 pad (bank-conflict avoidance)
#define XP 136   // LDS pitch (bf16) for x rows: 128 + 8 pad

static __device__ __forceinline__ bf16_t bf16_bits(unsigned short s) {
    union { unsigned short u; bf16_t b; } v; v.u = s; return v.b;
}
static __device__ __forceinline__ bf16_t to_bf16(float f) {
    union { float f; unsigned u; } v; v.f = f;
    unsigned r = v.u + 0x7FFFu + ((v.u >> 16) & 1u);   // round-to-nearest-even
    return bf16_bits((unsigned short)(r >> 16));
}
// A-fragment (16x32 bf16) per-lane load: row M = lane&15; 8 contiguous bf16 at
// col0, 8 more at col0+16 (lane>=16 uses col0 offset by +8, handled by caller).
static __device__ __forceinline__ v16bf load_a16(const bf16_t* rowp, int col0) {
    v8bf lo = *(const v8bf*)(rowp + col0);
    v8bf hi = *(const v8bf*)(rowp + col0 + 16);
    return __builtin_shufflevector(lo, hi, 0,1,2,3,4,5,6,7,8,9,10,11,12,13,14,15);
}
static __device__ __forceinline__ float sigmoid_fast(float x) {
    return 1.0f / (1.0f + __expf(-x));
}

// ---------------------------------------------------------------------------
// Pack W [F_IN x G4] and R [U_DIM x G4] (f32, row-major, row stride G4) into
// bf16 WMMA B-fragment order:
//   element index = ((nb*Kblocks + kc)*32 + lane)*16 + e
//   K = kc*32 + (lane>=16 ? 8 : 0) + (e&7) + (e>=8 ? 16 : 0),  N = nb*16 + (lane&15)
// so each lane's 16 B-values per fragment are 32 contiguous bytes.
// ---------------------------------------------------------------------------
__global__ void lstm_pack_wr_kernel(const float* __restrict__ W,
                                    const float* __restrict__ R,
                                    bf16_t* __restrict__ Wp,
                                    bf16_t* __restrict__ Rp) {
    const int WP_ELEMS = F_IN * G4;           // 131072
    const int RP_ELEMS = U_DIM * G4;          // 262144
    int tid = blockIdx.x * blockDim.x + threadIdx.x;
    if (tid >= WP_ELEMS + RP_ELEMS) return;
    bool isW = tid < WP_ELEMS;
    int idx = isW ? tid : tid - WP_ELEMS;
    int Kblocks = isW ? (F_IN / 32) : (U_DIM / 32);
    const float* src = isW ? W : R;
    bf16_t* dst = isW ? Wp : Rp;

    int e    = idx & 15;
    int lane = (idx >> 4) & 31;
    int blk  = idx >> 9;                       // nb*Kblocks + kc
    int kc   = blk % Kblocks;
    int nb   = blk / Kblocks;
    int K = kc * 32 + ((lane >> 4) << 3) + (e & 7) + ((e >> 3) << 4);
    int N = nb * 16 + (lane & 15);
    dst[idx] = to_bf16(src[(size_t)K * G4 + N]);
}

// ---------------------------------------------------------------------------
// Persistent recurrence kernel. grid = 8 groups, block = 512 threads (16 waves)
// ---------------------------------------------------------------------------
__launch_bounds__(512, 1)
__global__ void lstm_recurrence_kernel(const float* __restrict__ x,     // [B,T,F]
                                       const float* __restrict__ bias,  // [4U]
                                       const bf16_t* __restrict__ Wp,
                                       const bf16_t* __restrict__ Rp,
                                       float* __restrict__ out) {       // [B,U]
    __shared__ __align__(16) bf16_t hS[16 * HP];
    __shared__ __align__(16) bf16_t xS[16 * XP];

    const int tid   = threadIdx.x;
    const int lane  = tid & 31;
    const int wv    = __builtin_amdgcn_readfirstlane(tid >> 5); // wave-uniform unit block
    const int grp   = blockIdx.x;         // batch group (0..7)
    const int l15   = lane & 15;          // A row / C column within tile
    const int khalf = (lane >> 4) ? 8 : 0;

    // zero the h LDS region (h0 = 0)
    for (int i = tid; i < 16 * HP; i += 512) hS[i] = bf16_bits(0);

    // per-lane bias for each gate tile (same value for all 8 C rows)
    float bg[4];
#pragma unroll
    for (int gt = 0; gt < 4; ++gt) bg[gt] = bias[gt * U_DIM + wv * 16 + l15];

    // wave-uniform per-gate B base pointers (lane part added in the loop)
    const bf16_t* Wg[4];
    const bf16_t* Rg[4];
#pragma unroll
    for (int gt = 0; gt < 4; ++gt) {
        int nb = gt * 16 + wv;
        Wg[gt] = Wp + (size_t)nb * (4 * 512);   // (nb*4 + kc)*512
        Rg[gt] = Rp + (size_t)nb * (8 * 512);   // (nb*8 + kc)*512
    }
    const int laneB = lane * 16;                 // per-lane element offset in a fragment

    v8f c_frag; v8f h_frag;
#pragma unroll
    for (int r = 0; r < 8; ++r) { c_frag[r] = 0.0f; h_frag[r] = 0.0f; }

    // x staging: thread loads float4 of row m=(tid>>5), cols (tid&31)*4..+3
    const float* xstage = x + ((size_t)(grp * 16 + (tid >> 5)) * T_LEN) * F_IN + (tid & 31) * 4;
    bf16_t* xdst = &xS[(tid >> 5) * XP + (tid & 31) * 4];

    const bf16_t* xrowA = &xS[l15 * XP];
    const bf16_t* hrowA = &hS[l15 * HP];

    for (int t = 0; t < T_LEN; ++t) {
        // ---- stage x_t into LDS (bf16) ----
        f32x4 xv = *(const f32x4*)(xstage + (size_t)t * F_IN);
        xdst[0] = to_bf16(xv.x); xdst[1] = to_bf16(xv.y);
        xdst[2] = to_bf16(xv.z); xdst[3] = to_bf16(xv.w);
        // prefetch next timestep's x row into near caches while we compute
        __builtin_prefetch(xstage + (size_t)(t + 1) * F_IN, 0, 3);
        __syncthreads();   // x_t staged; previous h writes visible

        v8f acc[4];
#pragma unroll
        for (int gt = 0; gt < 4; ++gt)
#pragma unroll
            for (int r = 0; r < 8; ++r) acc[gt][r] = bg[gt];

        // ---- z += x_t @ W : K = 128 -> 4 chunks (do NOT unroll: keeps B loads
        //      per-iteration so they can't be hoisted across t and spilled) ----
#pragma unroll 1
        for (int kc = 0; kc < 4; ++kc) {
            v16bf a = load_a16(xrowA, kc * 32 + khalf);
            int off = kc * 512 + laneB;
#pragma unroll
            for (int gt = 0; gt < 4; ++gt) {
                const v16bf b = *(const v16bf*)(Wg[gt] + off);
                acc[gt] = __builtin_amdgcn_wmma_f32_16x16x32_bf16(
                    false, a, false, b, (short)0, acc[gt], false, false);
            }
        }
        // ---- z += h @ R : K = 256 -> 8 chunks ----
#pragma unroll 1
        for (int kc = 0; kc < 8; ++kc) {
            v16bf a = load_a16(hrowA, kc * 32 + khalf);
            int off = kc * 512 + laneB;
#pragma unroll
            for (int gt = 0; gt < 4; ++gt) {
                const v16bf b = *(const v16bf*)(Rg[gt] + off);
                acc[gt] = __builtin_amdgcn_wmma_f32_16x16x32_bf16(
                    false, a, false, b, (short)0, acc[gt], false, false);
            }
        }

        // ---- gates (i,f,o sigmoid; g linear), update c and h ----
#pragma unroll
        for (int r = 0; r < 8; ++r) {
            float si = sigmoid_fast(acc[0][r]);
            float sf = sigmoid_fast(acc[1][r]);
            float zg = acc[2][r];
            float so = sigmoid_fast(acc[3][r]);
            c_frag[r] = sf * c_frag[r] + si * zg;
            h_frag[r] = so * c_frag[r];
        }

        __syncthreads();   // all waves done reading old h from LDS

        // ---- write h tile back to LDS (C layout -> row-major bf16) ----
        int mbase = (lane >> 4) << 3;            // 0 or 8
#pragma unroll
        for (int r = 0; r < 8; ++r) {
            hS[(mbase + r) * HP + wv * 16 + l15] = to_bf16(h_frag[r]);
        }
    }

    __syncthreads();
    // ---- out = tanh(h_T) ----
    int mbase = (lane >> 4) << 3;
#pragma unroll
    for (int r = 0; r < 8; ++r) {
        int row  = grp * 16 + mbase + r;         // batch row
        int colu = wv * 16 + l15;                // unit
        out[(size_t)row * U_DIM + colu] = tanhf(h_frag[r]);
    }
}

// ---------------------------------------------------------------------------
extern "C" void kernel_launch(void* const* d_in, const int* in_sizes, int n_in,
                              void* d_out, int out_size, void* d_ws, size_t ws_size,
                              hipStream_t stream) {
    const float* x = (const float*)d_in[0];   // [128,1024,128]
    const float* W = (const float*)d_in[1];   // [128,1024]
    const float* R = (const float*)d_in[2];   // [256,1024]
    const float* b = (const float*)d_in[3];   // [1024]

    bf16_t* Wp = (bf16_t*)d_ws;               // 131072 bf16 = 256 KB
    bf16_t* Rp = Wp + (size_t)F_IN * G4;      // 262144 bf16 = 512 KB

    int packN = F_IN * G4 + U_DIM * G4;       // 393216
    lstm_pack_wr_kernel<<<(packN + 255) / 256, 256, 0, stream>>>(W, R, Wp, Rp);

    lstm_recurrence_kernel<<<B_TOT / 16, 512, 0, stream>>>(x, b, Wp, Rp, (float*)d_out);
}
